// VQ_5755256176822
// MI455X (gfx1250) — matrix operand order repaired and checked by reading, not compiled, not run on hardware
//
#include <hip/hip_runtime.h>

// ---------------------------------------------------------------------------
// VQ-VAE vector quantization forward for MI455X (gfx1250, wave32, WMMA + TDM)
//   inputs : encoder_inputs f32 [32,256,32,32], embedding_weight f32 [1024,256]
//   outputs: quantized f32 [32,256,32,32] ++ loss f32 [1]
// ---------------------------------------------------------------------------

typedef _Float16     v8h   __attribute__((ext_vector_type(8)));
typedef _Float16     v16h  __attribute__((ext_vector_type(16)));
typedef float        v8f   __attribute__((ext_vector_type(8)));
typedef unsigned int u32x4 __attribute__((ext_vector_type(4)));
typedef int          i32x4 __attribute__((ext_vector_type(4)));
typedef int          i32x8 __attribute__((ext_vector_type(8)));

#define VQ_B   32
#define VQ_D   256
#define VQ_HW  1024            // 32*32
#define VQ_N   32768           // B*HW
#define VQ_K   1024            // codes
#define VQ_ND  8388608         // N*D elements of quantized output

// LDS code-tile layout: 16 rows x 256 f16, row stride padded 512B -> 528B
// (132 dwords; 132 % 64 = 4 -> 16 columns land on distinct LDS banks)
#define LDSROW 264             // f16 elements per padded row

// ---------------------------------------------------------------------------
// Kernel 1: codebook -> f16 copy + bias[n] = -0.5*||e_n||^2
// ---------------------------------------------------------------------------
__global__ __launch_bounds__(256) void vq_prep_codebook(
    const float* __restrict__ E, _Float16* __restrict__ eh,
    float* __restrict__ bias) {
  const int n = blockIdx.x;
  const int d = threadIdx.x;
  float v = E[n * VQ_D + d];
  eh[n * VQ_D + d] = (_Float16)v;
  __shared__ float red[256];
  red[d] = v * v;
  __syncthreads();
  #pragma unroll
  for (int s = 128; s > 0; s >>= 1) {
    if (d < s) red[d] += red[d + s];
    __syncthreads();
  }
  if (d == 0) bias[n] = -0.5f * red[0];
}

// ---------------------------------------------------------------------------
// Kernel 2: BCHW f32 -> BHWC f16 (LDS-tiled transpose), xh[n][c], n=b*HW+hw
// ---------------------------------------------------------------------------
__global__ __launch_bounds__(256) void vq_transpose_convert(
    const float* __restrict__ in, _Float16* __restrict__ xh) {
  const int b   = blockIdx.z;
  const int hw0 = blockIdx.x * 32;
  const int c0  = blockIdx.y * 32;
  __shared__ float tile[32][33];
  #pragma unroll
  for (int i = 0; i < 4; ++i) {
    int c = c0 + threadIdx.y + 8 * i;
    tile[threadIdx.y + 8 * i][threadIdx.x] =
        in[((size_t)(b * VQ_D + c)) * VQ_HW + hw0 + threadIdx.x];
  }
  __syncthreads();
  #pragma unroll
  for (int i = 0; i < 4; ++i) {
    int hw = hw0 + threadIdx.y + 8 * i;
    xh[((size_t)(b * VQ_HW + hw)) * VQ_D + c0 + threadIdx.x] =
        (_Float16)tile[threadIdx.x][threadIdx.y + 8 * i];
  }
}

// ---------------------------------------------------------------------------
// Tile staging: TDM (tensor_load_to_lds) when available, else cooperative copy
// ---------------------------------------------------------------------------
#if __has_builtin(__builtin_amdgcn_tensor_load_to_lds)
#define VQ_USE_TDM 1
#else
#define VQ_USE_TDM 0
#endif

#if VQ_USE_TDM
// One contiguous 8KB tile (16 codes x 512B), moved as 1024 x 8B units with
// TDM LDS padding: +16B every 512B  =>  padded 528B LDS row stride.
static __device__ inline void vq_tdm_load_tile(const _Float16* eh,
                                               unsigned lds_addr, int t) {
  unsigned long long ga =
      (unsigned long long)(uintptr_t)(eh + (size_t)t * 16 * VQ_D);
  u32x4 g0;
  g0[0] = 1u;                                              // count=1 (valid)
  g0[1] = lds_addr;                                        // LDS dest (bytes)
  g0[2] = (unsigned)ga;                                    // global_addr[31:0]
  g0[3] = (unsigned)((ga >> 32) & 0x01FFFFFFu) | (2u << 30); // [56:32] | type=2
  i32x8 g1;
  g1[0] = (int)((3u << 16) |        // data_size = 8B
                (1u << 20) |        // pad_enable
                (6u << 22) |        // pad_interval: 128 dwords (512B)
                (3u << 25));        // pad_amount:   4 dwords  (16B)
  g1[1] = 0;                        // no atomic barrier; tensor_dim0[15:0]=0
  g1[2] = (int)((1u << 16) | 1u);   // tensor_dim0=65536 (hi=1), tensor_dim1=1
  g1[3] = (int)(1024u << 16);       // tile_dim0=1024 (8B units)
  g1[4] = 0;                        // tile_dim1/2 unused (1D transfer)
  g1[5] = 65536;                    // tensor_dim0_stride (unused for 1D)
  g1[6] = 0;
  g1[7] = 0;
  i32x4 z4 = {0, 0, 0, 0};
#if defined(__clang_major__) && __clang_major__ >= 23
  i32x8 z8 = {0, 0, 0, 0, 0, 0, 0, 0};
  __builtin_amdgcn_tensor_load_to_lds(g0, g1, z4, z4, z8, 0);
#else
  __builtin_amdgcn_tensor_load_to_lds(g0, g1, z4, z4, 0);
#endif
}
#endif

// ---------------------------------------------------------------------------
// Kernel 3: WMMA distance argmin with double-buffered LDS code tiles.
//   score[m][n] = x_m . e_n - 0.5*||e_n||^2 ; argmax over n == argmin dist.
//   4 waves / block; each wave owns 16 rows; block shares B tiles via LDS.
// ---------------------------------------------------------------------------
__global__ __launch_bounds__(128) void vq_argmin(
    const _Float16* __restrict__ xh, const _Float16* __restrict__ eh,
    const float* __restrict__ bias, int* __restrict__ idxOut) {
  __shared__ __align__(128) _Float16 ebuf[2][16 * LDSROW];   // 2 x 8448 B

  const int tid  = threadIdx.x;
  const int lane = tid & 31;
  const int wave = tid >> 5;
  const int rowBase = (blockIdx.x * 4 + wave) * 16;
  const int m = lane & 15;   // A row / C column / B column index within tile
  const int h = lane >> 4;   // lane half selects K sub-block

  // ---- load A tile: lane holds row (rowBase+m), K-perm per CDNA5 A layout:
  //      elems[0..7]=K(8h..8h+7), elems[8..15]=K(16+8h..16+8h+7) per 32-step
  const _Float16* xrow = xh + (size_t)(rowBase + m) * VQ_D;
  v16h A[8];
  #pragma unroll
  for (int ks = 0; ks < 8; ++ks) {
    v8h lo = *(const v8h*)(xrow + 32 * ks + 8 * h);
    v8h hi = *(const v8h*)(xrow + 32 * ks + 16 + 8 * h);
    #pragma unroll
    for (int j = 0; j < 8; ++j) { A[ks][j] = lo[j]; A[ks][j + 8] = hi[j]; }
  }

  float bestScore[8];
  int   bestIdx[8];
  #pragma unroll
  for (int v = 0; v < 8; ++v) { bestScore[v] = -3.0e38f; bestIdx[v] = 0; }

  // ---- stage tile 0
#if VQ_USE_TDM
  if (wave == 0) {
    vq_tdm_load_tile(eh, (unsigned)(uintptr_t)&ebuf[0][0], 0);
    __builtin_amdgcn_s_wait_tensorcnt(0);
  }
#else
  {
    #pragma unroll
    for (int i = 0; i < 4; ++i) {
      int chunk = tid + i * 128;          // 512 x 16B chunks per 8KB tile
      int row = chunk >> 5;
      int col = chunk & 31;
      *(v8h*)(&ebuf[0][row * LDSROW + col * 8]) =
          *(const v8h*)(eh + (size_t)row * VQ_D + col * 8);
    }
  }
#endif
  __syncthreads();

  int buf = 0;
  for (int t = 0; t < 64; ++t) {
    const int nb = buf ^ 1;
#if VQ_USE_TDM
    if (wave == 0 && t + 1 < 64)
      vq_tdm_load_tile(eh, (unsigned)(uintptr_t)&ebuf[nb][0], t + 1);
#else
    if (t + 1 < 64) {
      #pragma unroll
      for (int i = 0; i < 4; ++i) {
        int chunk = tid + i * 128;
        int row = chunk >> 5;
        int col = chunk & 31;
        *(v8h*)(&ebuf[nb][row * LDSROW + col * 8]) =
            *(const v8h*)(eh + ((size_t)(t + 1) * 16 + row) * VQ_D + col * 8);
      }
    }
#endif

    // ---- compute on current tile (B from LDS)
    const int n = t * 16 + m;                 // this lane's code column
    const float bs = bias[n];
    v8f c;
    #pragma unroll
    for (int j = 0; j < 8; ++j) c[j] = bs;    // C init = -0.5*||e_n||^2

    const _Float16* ebt = &ebuf[buf][0];
    // B layout: lane holds column n, elems[0..7]=K(8h..8h+7), elems[8..15]=K(16+8h..16+8h+7) per 32-step
    #pragma unroll
    for (int ks = 0; ks < 8; ++ks) {
      const _Float16* bp = ebt + m * LDSROW + 32 * ks + 8 * h;
      v8h blo = *(const v8h*)(bp);
      v8h bhi = *(const v8h*)(bp + 16);
      v16h bv;
      #pragma unroll
      for (int j = 0; j < 8; ++j) { bv[j] = blo[j]; bv[j + 8] = bhi[j]; }
      c = __builtin_amdgcn_wmma_f32_16x16x32_f16(
          /*neg_a=*/false, A[ks], /*neg_b=*/false, bv,
          /*c_mod=*/(short)0, c, /*reuse_a=*/false, /*reuse_b=*/false);
    }

    // running per-lane argmax; C vgpr v holds (row = v + 8h, col = n)
    #pragma unroll
    for (int v = 0; v < 8; ++v) {
      if (c[v] > bestScore[v]) { bestScore[v] = c[v]; bestIdx[v] = n; }
    }

#if VQ_USE_TDM
    if (wave == 0 && t + 1 < 64) __builtin_amdgcn_s_wait_tensorcnt(0);
#endif
    __syncthreads();
    buf = nb;
  }

  // cross-lane argmax within each 16-lane half (rows are shared across the
  // half; columns differ). Tie-break toward smaller code index (argmin-first).
  #pragma unroll
  for (int v = 0; v < 8; ++v) {
    float s = bestScore[v];
    int   bi = bestIdx[v];
    #pragma unroll
    for (int d = 1; d < 16; d <<= 1) {
      float s2 = __shfl_xor(s, d, 32);
      int   i2 = __shfl_xor(bi, d, 32);
      if (s2 > s || (s2 == s && i2 < bi)) { s = s2; bi = i2; }
    }
    if (m == 0) idxOut[rowBase + 8 * h + v] = bi;
  }
}

// ---------------------------------------------------------------------------
// Kernel 4: gather E[idx] -> BCHW output + per-block partial sum of (q-x)^2
// ---------------------------------------------------------------------------
__global__ __launch_bounds__(256) void vq_gather(
    const float* __restrict__ in, const float* __restrict__ E,
    const int* __restrict__ idx, float* __restrict__ out,
    float* __restrict__ partial) {
  const int c = blockIdx.x;
  const int b = blockIdx.y;
  const float* inrow  = in  + ((size_t)(b * VQ_D + c)) * VQ_HW;
  float*       outrow = out + ((size_t)(b * VQ_D + c)) * VQ_HW;
  const int*   idxb   = idx + b * VQ_HW;
  float acc = 0.0f;
  #pragma unroll
  for (int i = 0; i < 4; ++i) {
    int hw = i * 256 + threadIdx.x;
    float q = E[(size_t)idxb[hw] * VQ_D + c];
    float x = inrow[hw];
    float d = q - x;
    acc += d * d;
    outrow[hw] = q;            // straight-through forward value == q
  }
  __shared__ float red[256];
  red[threadIdx.x] = acc;
  __syncthreads();
  #pragma unroll
  for (int s = 128; s > 0; s >>= 1) {
    if (threadIdx.x < s) red[threadIdx.x] += red[threadIdx.x + s];
    __syncthreads();
  }
  if (threadIdx.x == 0) partial[b * VQ_D + c] = red[0];
}

// ---------------------------------------------------------------------------
// Kernel 5: deterministic final reduction -> loss = 1.25 * mean((q-x)^2)
// ---------------------------------------------------------------------------
__global__ __launch_bounds__(256) void vq_loss_final(
    const float* __restrict__ partial, float* __restrict__ out) {
  float acc = 0.0f;
  for (int i = threadIdx.x; i < VQ_B * VQ_D; i += 256) acc += partial[i];
  __shared__ float red[256];
  red[threadIdx.x] = acc;
  __syncthreads();
  #pragma unroll
  for (int s = 128; s > 0; s >>= 1) {
    if (threadIdx.x < s) red[threadIdx.x] += red[threadIdx.x + s];
    __syncthreads();
  }
  if (threadIdx.x == 0) out[VQ_ND] = red[0] * (1.25f / (float)VQ_ND);
}

// ---------------------------------------------------------------------------
// Launcher
// ---------------------------------------------------------------------------
extern "C" void kernel_launch(void* const* d_in, const int* in_sizes, int n_in,
                              void* d_out, int out_size, void* d_ws, size_t ws_size,
                              hipStream_t stream) {
  const float* enc = (const float*)d_in[0];   // [32,256,32,32]
  const float* E   = (const float*)d_in[1];   // [1024,256]
  float* out = (float*)d_out;                 // [8388608] quantized + [1] loss

  // Workspace layout (ws): idx | eh | bias | partial   (~0.7 MB total)
  char* ws = (char*)d_ws;
  int*      idx     = (int*)(ws);                         // 32768 * 4   = 128 KB
  _Float16* eh      = (_Float16*)(ws + (128 << 10));      // 1024*256*2  = 512 KB
  float*    bias    = (float*)(ws + (640 << 10));         // 1024 * 4    =   4 KB
  float*    partial = (float*)(ws + (648 << 10));         // 8192 * 4    =  32 KB

  // Stage f16 activations in the front 16 MB of d_out (fully overwritten by
  // vq_gather afterwards, so this is safe and deterministic).
  _Float16* xh = (_Float16*)d_out;                        // 32768*256*2 = 16 MB

  vq_prep_codebook<<<VQ_K, 256, 0, stream>>>(E, eh, bias);
  {
    dim3 grid(VQ_HW / 32, VQ_D / 32, VQ_B);
    dim3 block(32, 8);
    vq_transpose_convert<<<grid, block, 0, stream>>>(enc, xh);
  }
  vq_argmin<<<VQ_N / 64, 128, 0, stream>>>(xh, eh, bias, idx);
  {
    dim3 grid(VQ_D, VQ_B);
    vq_gather<<<grid, 256, 0, stream>>>(enc, E, idx, out, partial);
  }
  vq_loss_final<<<1, 256, 0, stream>>>(partial, out);
}
